// RankPooling_69827578298759
// MI455X (gfx1250) — matrix operand order repaired and checked by reading, not compiled, not run on hardware
//
#include <hip/hip_runtime.h>

typedef __attribute__((ext_vector_type(16))) _Float16 v16h;
typedef __attribute__((ext_vector_type(8)))  _Float16 h8;
typedef __attribute__((ext_vector_type(8)))  float    v8f;
typedef __attribute__((ext_vector_type(2)))  float    v2f;

#define Bn 256
#define Dn 256
#define Tn 512
#define FSTR 520                      // halves per F row: 512 + 8 pad -> 1040 B (≡16 mod 256: no bank conflicts)
#define GSTR 260                      // floats per G row: 256 + 4 pad -> 1040 B
#define SM_MAIN (Dn * FSTR * 2)       // 266240 B, shared between f16 F and f32 G (time-multiplexed)
#define SM_TOTAL (SM_MAIN + Tn * 4 + Dn * 4 + Dn * 4)   // + rnorm + rhs + x = 270336 B < 320 KB

__global__ __launch_bounds__(512)
void rankpool_fused_kernel(const float* __restrict__ inp, float* __restrict__ out)
{
    const int tid = threadIdx.x;
    const int b   = blockIdx.x;

    extern __shared__ char smem[];
    _Float16* Fsm   = (_Float16*)smem;                 // [Dn][FSTR] f16, phases 1-4
    float*    Gsm   = (float*)smem;                    // [Dn][GSTR] f32, phases 4b-5 (reuses F bytes)
    float*    rnorm = (float*)(smem + SM_MAIN);        // [Tn]
    float*    rhs   = rnorm + Tn;                      // [Dn]
    float*    xv    = rhs + Dn;                        // [Dn]

    const int wave = tid >> 5;
    const int lane = tid & 31;
    const int lhi  = lane >> 4;       // 0 / 1 (half-wave select)
    const int llo  = lane & 15;

    // ---- Phase 1: cumulative mean over t, store f16 into LDS (thread = feature row) ----
    if (tid < Dn) {
        const float* row  = inp + ((size_t)b * Dn + tid) * Tn;
        _Float16*    frow = Fsm + (size_t)tid * FSTR;
        float run = 0.f;
        for (int t = 0; t < Tn; t += 4) {
            if (t + 128 < Tn) __builtin_prefetch(row + t + 128, 0, 0);
            float4 v = *(const float4*)(row + t);
            run += v.x; frow[t + 0] = (_Float16)(run / (float)(t + 1));
            run += v.y; frow[t + 1] = (_Float16)(run / (float)(t + 2));
            run += v.z; frow[t + 2] = (_Float16)(run / (float)(t + 3));
            run += v.w; frow[t + 3] = (_Float16)(run / (float)(t + 4));
        }
    }
    __syncthreads();

    // ---- Phase 2: per-time-column inverse L2 norm over features (thread = time column) ----
    {
        const int t = tid;            // 512 threads <-> Tn columns
        float s = 0.f;
        for (int d = 0; d < Dn; ++d) {
            float c = (float)Fsm[(size_t)d * FSTR + t];
            s += c * c;
        }
        rnorm[t] = rsqrtf(s);
    }
    __syncthreads();

    // ---- Phase 3: normalize F in place; rhs[d] = sum_t f[d][t] * (t+1) ----
    if (tid < Dn) {
        _Float16* frow = Fsm + (size_t)tid * FSTR;
        float acc = 0.f;
        for (int t = 0; t < Tn; ++t) {
            float f = (float)frow[t] * rnorm[t];
            frow[t] = (_Float16)f;
            acc += f * (float)(t + 1);
        }
        rhs[tid] = acc;
    }
    __syncthreads();

    // ---- Phase 4: Gram G = F * F^T via v_wmma_f32_16x16x32_f16 ----
    // 16 waves; wave w owns C row-block w (16 tiles of 16x16, v8f each).
    {
        v8f acc[16] = {};                 // 128 VGPRs of f32 accumulators
        const int mRow = wave * 16 + llo; // A-matrix row for this lane

        for (int k0 = 0; k0 < Tn; k0 += 32) {
            v16h a;
            {
                const _Float16* ar = Fsm + (size_t)mRow * FSTR;
                h8 lo = *(const h8*)(ar + k0 + lhi * 8);
                h8 hi = *(const h8*)(ar + k0 + 16 + lhi * 8);
                #pragma unroll
                for (int i = 0; i < 8; ++i) { a[i] = lo[i]; a[8 + i] = hi[i]; }
            }
            #pragma unroll
            for (int n = 0; n < 16; ++n) {
                const _Float16* br = Fsm + (size_t)(n * 16 + llo) * FSTR + k0 + lhi * 8;
                h8 b0 = *(const h8*)(br);
                h8 b1 = *(const h8*)(br + 16);
                v16h bm;
                #pragma unroll
                for (int i = 0; i < 8; ++i) { bm[i] = b0[i]; bm[8 + i] = b1[i]; }
                acc[n] = __builtin_amdgcn_wmma_f32_16x16x32_f16(
                    false, a, false, bm, (short)0, acc[n], false, false);
            }
        }
        __syncthreads();   // all WMMA reads of F done before G overwrites the region

        // Scatter C tiles into G (f32). C/D layout: VGPR r -> M = r + 8*lhi, N = llo.
        #pragma unroll
        for (int n = 0; n < 16; ++n) {
            float* gtile = Gsm + (size_t)(wave * 16) * GSTR + n * 16;
            #pragma unroll
            for (int r = 0; r < 8; ++r)
                gtile[(size_t)(r + lhi * 8) * GSTR + llo] = acc[n][r];
        }
    }
    __syncthreads();

    // ---- Phase 5: blocked LU (NB=16, no pivoting; G is SPD) with fp32-WMMA trailing update ----
    for (int kb = 0; kb < 16; ++kb) {
        const int c0 = kb * 16;

        // (a) Panel factorization over the 16 block columns; rhs forward-substitution fused.
        for (int k = c0; k < c0 + 16; ++k) {
            if (tid > k && tid < Dn) {
                float f = Gsm[(size_t)tid * GSTR + k] / Gsm[(size_t)k * GSTR + k];
                Gsm[(size_t)tid * GSTR + k] = f;                     // store L multiplier
                for (int j = k + 1; j < c0 + 16; ++j)
                    Gsm[(size_t)tid * GSTR + j] -= f * Gsm[(size_t)k * GSTR + j];
                rhs[tid] -= f * rhs[k];
            }
            __syncthreads();
        }

        // (b) U row-block: columns right of the panel; one thread per column, barrier-free.
        if (kb < 15) {
            const int j = c0 + 16 + tid;
            if (j < Dn) {
                for (int r = c0 + 1; r < c0 + 16; ++r) {
                    float val = Gsm[(size_t)r * GSTR + j];
                    for (int m = c0; m < r; ++m)
                        val -= Gsm[(size_t)r * GSTR + m] * Gsm[(size_t)m * GSTR + j];
                    Gsm[(size_t)r * GSTR + j] = val;
                }
            }
            __syncthreads();

            // (c) Trailing Schur update A -= L*U via v_wmma_f32_16x16x4_f32 (K=16 -> 4 steps).
            const int nt = 15 - kb;             // trailing 16x16 tiles per dimension
            const int r0 = c0 + 16;
            for (int t = wave; t < nt * nt; t += 16) {
                const int i0 = r0 + (t / nt) * 16;
                const int j0 = r0 + (t % nt) * 16;
                v8f c;
                #pragma unroll
                for (int r = 0; r < 8; ++r)
                    c[r] = Gsm[(size_t)(i0 + r + lhi * 8) * GSTR + j0 + llo];
                #pragma unroll
                for (int kk = 0; kk < 4; ++kk) {
                    const int kb4 = c0 + kk * 4;
                    // A (16x4 f32): lane row = i0+llo; VGPR0 = K{0|2}, VGPR1 = K{1|3}. Load -L.
                    v2f a;
                    a[0] = -Gsm[(size_t)(i0 + llo) * GSTR + kb4 + 2 * lhi];
                    a[1] = -Gsm[(size_t)(i0 + llo) * GSTR + kb4 + 2 * lhi + 1];
                    // B (4x16 f32): lane col = j0+llo; VGPR0 = row{0|1}, VGPR1 = row{2|3}.
                    v2f bv;
                    bv[0] = Gsm[(size_t)(kb4 + lhi) * GSTR + j0 + llo];
                    bv[1] = Gsm[(size_t)(kb4 + 2 + lhi) * GSTR + j0 + llo];
                    c = __builtin_amdgcn_wmma_f32_16x16x4_f32(
                        false, a, false, bv, (short)0, c, false, false);
                }
                #pragma unroll
                for (int r = 0; r < 8; ++r)
                    Gsm[(size_t)(i0 + r + lhi * 8) * GSTR + j0 + llo] = c[r];
            }
            __syncthreads();
        }
    }

    // ---- Phase 6: blocked back-substitution (U x = rhs) ----
    for (int bb = 15; bb >= 0; --bb) {
        const int i0 = bb * 16;
        // 16x16 upper-triangular solve, wave-synchronous in wave 0 (shfl broadcast, no barriers).
        if (wave == 0) {
            float val = (lane < 16) ? rhs[i0 + lane] : 0.f;
            for (int k = 15; k >= 0; --k) {
                float xk = val / Gsm[(size_t)(i0 + k) * GSTR + i0 + k];
                xk = __shfl(xk, k, 32);          // take lane k's value
                if (lane == k) xv[i0 + k] = xk;
                if (lane < k)
                    val -= Gsm[(size_t)(i0 + lane) * GSTR + (i0 + k)] * xk;
            }
        }
        __syncthreads();
        // Rank-16 update of the rows above this block (one thread per row, barrier-free).
        if (tid < i0) {
            float s = 0.f;
            #pragma unroll
            for (int m = 0; m < 16; ++m)
                s += Gsm[(size_t)tid * GSTR + i0 + m] * xv[i0 + m];
            rhs[tid] -= s;
        }
        __syncthreads();
    }

    // ---- Output: concat(w, w) ----
    if (tid < Dn) {
        float w = xv[tid];
        out[(size_t)b * (2 * Dn) + tid]      = w;
        out[(size_t)b * (2 * Dn) + Dn + tid] = w;
    }
}

extern "C" void kernel_launch(void* const* d_in, const int* in_sizes, int n_in,
                              void* d_out, int out_size, void* d_ws, size_t ws_size,
                              hipStream_t stream) {
    (void)in_sizes; (void)n_in; (void)d_ws; (void)ws_size; (void)out_size;
    const float* inp = (const float*)d_in[0];
    float*       out = (float*)d_out;
    rankpool_fused_kernel<<<dim3(Bn), dim3(512), SM_TOTAL, stream>>>(inp, out);
}
